// TransformerDecoder_68693706932485
// MI455X (gfx1250) — compile-verified, matrix operand order checked by reading, not run on hardware
//
#include <hip/hip_runtime.h>
#include <hip/hip_bf16.h>
#include <stdint.h>

typedef __attribute__((ext_vector_type(16))) _Float16 v16h;
typedef __attribute__((ext_vector_type(8)))  _Float16 v8h;
typedef __attribute__((ext_vector_type(4)))  _Float16 v4h;
typedef __attribute__((ext_vector_type(8)))  float    v8f;
typedef __attribute__((ext_vector_type(4)))  float    fx4;

__device__ __forceinline__ v8f wmma16(const v16h& a, const v16h& b, const v8f& c) {
  // D = A(16x32 f16) * B(32x16 f16) + C(16x16 f32)
  return __builtin_amdgcn_wmma_f32_16x16x32_f16(
      /*neg_a=*/false, a, /*neg_b=*/false, b,
      /*c_mod=*/(short)0, c, /*reuse_a=*/false, /*reuse_b=*/false);
}

// 16-byte async global->LDS copy (ASYNCcnt-tracked, CDNA5 TDM-adjacent path)
__device__ __forceinline__ void async_cp16(uint32_t lds_addr, const void* gaddr) {
  asm volatile("global_load_async_to_lds_b128 %0, %1, off"
               :: "v"(lds_addr), "v"(gaddr) : "memory");
}
__device__ __forceinline__ void wait_async0() {
  asm volatile("s_wait_asynccnt 0x0" ::: "memory");
}

// ---------------------------------------------------------------------------
// Tiled WMMA GEMM:  C[M,N] = act((A[M,K] @ B[K,N] + bias[N]) * oscale)
// BM=128 BN=128 BK=32; 256 threads = 8 waves; wave -> 64x32 patch (4x2 tiles).
// B is pre-converted f16. A is f32 (converted in staging) or f16 (staged with
// global_load_async_to_lds_b128 -- raw copy, no VALU). LDS double-buffered,
// software-pipelined. Requires N%128==0, K%32==0; AF16 path assumes M%128==0.
// ---------------------------------------------------------------------------
template <bool AF16, bool OF16>
__global__ __launch_bounds__(256) void gemm_wmma_kernel(
    const void* __restrict__ Av, const _Float16* __restrict__ B,
    const float* __restrict__ bias, void* __restrict__ Cv,
    int M, int N, int K, int relu, float oscale)
{
  const float*    Af = (const float*)Av;
  const _Float16* Ah = (const _Float16*)Av;
  float*          Cf = (float*)Cv;
  _Float16*       Ch = (_Float16*)Cv;

  __shared__ __align__(16) _Float16 As[2][128][40];  // [buf][m][k]
  __shared__ __align__(16) _Float16 Bs[2][128][40];  // [buf][n][k] (B transposed)

  const int tid   = threadIdx.x;
  const int lane  = tid & 31;
  const int wave  = tid >> 5;
  const int wm    = wave >> 2;        // 0..1 : 64-row half
  const int wn    = wave & 3;         // 0..3 : 32-col strip
  const int tileM = blockIdx.y * 128;
  const int tileN = blockIdx.x * 128;
  const int rlo   = lane & 15;
  const int kb    = (lane >> 4) * 8;  // fragment K-base per ISA layout
  const bool fullM = (tileM + 128 <= M);

  // B staging: thread owns column n = tid&127, half-range of k (16 values)
  const int bn  = tid & 127;
  const int bk0 = (tid >> 7) * 16;

  v8f acc[4][2];
#pragma unroll
  for (int i = 0; i < 4; ++i)
#pragma unroll
    for (int j = 0; j < 2; ++j) acc[i][j] = {};

  // f16-A tile staging: 512 x 16B units, 2 per thread, async copy to LDS
  auto asyncA = [&](int kofs, int buf) {
#pragma unroll
    for (int i = 0; i < 2; ++i) {
      const int u  = i * 256 + tid;
      const int r  = u >> 2, c8 = u & 3;
      const uint32_t lds = (uint32_t)(uintptr_t)&As[buf][r][c8 * 8];
      async_cp16(lds, (const void*)(Ah + (size_t)(tileM + r) * K + kofs + c8 * 8));
    }
  };

  // ---- prologue: stage k0 = 0 into buffer 0 ----
  {
    if constexpr (AF16) {
      asyncA(0, 0);
    } else {
      fx4 af[4];
#pragma unroll
      for (int i = 0; i < 4; ++i) {
        const int idx = i * 256 + tid;
        const int r = idx >> 3, c4 = idx & 7;
        af[i] = {};
        if (fullM || (tileM + r) < M)
          af[i] = *(const fx4*)(Af + (size_t)(tileM + r) * K + c4 * 4);
      }
#pragma unroll
      for (int i = 0; i < 4; ++i) {
        const int idx = i * 256 + tid;
        const int r = idx >> 3, c4 = idx & 7;
        *(v4h*)&As[0][r][c4 * 4] = __builtin_convertvector(af[i], v4h);
      }
    }
    _Float16 bf[16];
#pragma unroll
    for (int i = 0; i < 16; ++i)
      bf[i] = B[(size_t)(bk0 + i) * N + (tileN + bn)];
    v8h h0, h1;
#pragma unroll
    for (int i = 0; i < 8; ++i) { h0[i] = bf[i]; h1[i] = bf[8 + i]; }
    *(v8h*)&Bs[0][bn][bk0]     = h0;
    *(v8h*)&Bs[0][bn][bk0 + 8] = h1;
    if constexpr (AF16) wait_async0();
  }
  __syncthreads();

  int cur = 0;
  for (int k0 = 0; k0 < K; k0 += 32) {
    const int nk = k0 + 32;
    const bool nx = (nk < K);

    // ---- issue next tile's loads (latency hidden under WMMA) ----
    fx4      aregs[4];
    _Float16 bregs[16];
    if (nx) {
      if constexpr (AF16) {
        asyncA(nk, cur ^ 1);
      } else {
#pragma unroll
        for (int i = 0; i < 4; ++i) {
          const int idx = i * 256 + tid;
          const int r = idx >> 3, c4 = idx & 7;
          aregs[i] = {};
          if (fullM || (tileM + r) < M)
            aregs[i] = *(const fx4*)(Af + (size_t)(tileM + r) * K + nk + c4 * 4);
        }
      }
#pragma unroll
      for (int i = 0; i < 16; ++i)
        bregs[i] = B[(size_t)(nk + bk0 + i) * N + (tileN + bn)];
      if (nk + 32 < K)  // prefetch k+64 B tile (global_prefetch_b8)
        __builtin_prefetch(B + (size_t)(nk + 32 + bk0) * N + tileN + bn, 0, 1);
    }

    // ---- compute on current buffer ----
#pragma unroll
    for (int mt = 0; mt < 4; ++mt) {
      const int ar = wm * 64 + mt * 16 + rlo;
      v16h a;
#pragma unroll
      for (int i = 0; i < 8; ++i) { a[i] = As[cur][ar][kb + i]; a[8 + i] = As[cur][ar][kb + 16 + i]; }
#pragma unroll
      for (int nt = 0; nt < 2; ++nt) {
        const int bc = wn * 32 + nt * 16 + rlo;
        v16h b;
#pragma unroll
        for (int i = 0; i < 8; ++i) { b[i] = Bs[cur][bc][kb + i]; b[8 + i] = Bs[cur][bc][kb + 16 + i]; }
        acc[mt][nt] = wmma16(a, b, acc[mt][nt]);
      }
    }

    // ---- store next tile into the other buffer ----
    if (nx) {
      if constexpr (!AF16) {
#pragma unroll
        for (int i = 0; i < 4; ++i) {
          const int idx = i * 256 + tid;
          const int r = idx >> 3, c4 = idx & 7;
          *(v4h*)&As[cur ^ 1][r][c4 * 4] = __builtin_convertvector(aregs[i], v4h);
        }
      }
      v8h h0, h1;
#pragma unroll
      for (int i = 0; i < 8; ++i) { h0[i] = bregs[i]; h1[i] = bregs[8 + i]; }
      *(v8h*)&Bs[cur ^ 1][bn][bk0]     = h0;
      *(v8h*)&Bs[cur ^ 1][bn][bk0 + 8] = h1;
    }
    if constexpr (AF16) wait_async0();
    __syncthreads();
    cur ^= 1;
  }

  // epilogue: bias, oscale (+ReLU); C-layout: row r + 8*(lane/16), col lane%16
#pragma unroll
  for (int mt = 0; mt < 4; ++mt)
#pragma unroll
    for (int nt = 0; nt < 2; ++nt)
#pragma unroll
      for (int r = 0; r < 8; ++r) {
        const int grow = tileM + wm * 64 + mt * 16 + r + 8 * (lane >> 4);
        const int gcol = tileN + wn * 32 + nt * 16 + rlo;
        if (grow < M) {
          float v = (acc[mt][nt][r] + bias[gcol]) * oscale;
          if (relu) v = fmaxf(v, 0.0f);
          if constexpr (OF16) Ch[(size_t)grow * N + gcol] = (_Float16)v;
          else                Cf[(size_t)grow * N + gcol] = v;
        }
      }
}

// ---------------------------------------------------------------------------
// Fused causal flash attention, head_dim=64, T=512, D=1024 (16 heads packed).
// f16 Q/K/V in (Q pre-scaled by 1/8 in its projection GEMM), f16 out.
// Block = 128 threads (4 waves); block covers 64 query rows of one (n,h).
// K tile staged via global_load_async_to_lds_b128; V register-transposed.
// ---------------------------------------------------------------------------
__global__ __launch_bounds__(128) void attn_kernel(
    const _Float16* __restrict__ Q, const _Float16* __restrict__ K,
    const _Float16* __restrict__ V, _Float16* __restrict__ O)
{
  const int T = 512, D = 1024, HD = 64;
  __shared__ __align__(16) _Float16 Ks[32][72];     // [key][dim]
  __shared__ __align__(16) _Float16 VsT[64][40];    // [dim][key]
  __shared__ __align__(16) _Float16 Ps[4][16][40];  // per-wave P staging

  const int tid   = threadIdx.x;
  const int lane  = tid & 31;
  const int w     = tid >> 5;           // wave 0..3 -> 16 query rows each
  const int n     = blockIdx.z;
  const int h     = blockIdx.y;
  const int qbase = blockIdx.x * 64;
  const int rlo   = lane & 15;
  const int kb    = (lane >> 4) * 8;
  const int qrow  = qbase + w * 16 + rlo;

  // Q fragments (already scaled by 1/sqrt(64) in the projection epilogue)
  v16h qa[2];
#pragma unroll
  for (int c = 0; c < 2; ++c) {
    const _Float16* qp = Q + (size_t)(n * T + qrow) * D + h * HD + c * 32;
    v8h q0 = *(const v8h*)(qp + kb);
    v8h q1 = *(const v8h*)(qp + kb + 16);
#pragma unroll
    for (int i = 0; i < 8; ++i) { qa[c][i] = q0[i]; qa[c][8 + i] = q1[i]; }
  }

  float mrow[8], lrow[8];
#pragma unroll
  for (int r = 0; r < 8; ++r) { mrow[r] = -1e30f; lrow[r] = 0.0f; }
  v8f accO[4];
#pragma unroll
  for (int t = 0; t < 4; ++t) accO[t] = {};

  const int ntiles = (qbase >> 5) + 2;  // keys needed: [0, qbase+64)
  for (int j = 0; j < ntiles; ++j) {
    const int kb0 = j * 32;

    // K tile (32 keys x 64 dims): async raw copy, 128 x 16B units, 1/thread
    {
      const int key = tid >> 2, d8 = tid & 3;
      const uint32_t lds = (uint32_t)(uintptr_t)&Ks[key][d8 * 8];
      async_cp16(lds, (const void*)(K + (size_t)(n * T + kb0 + key) * D + h * HD + d8 * 8));
    }
    // V tile: vectorized loads + register transpose into VsT[dim][key]
#pragma unroll
    for (int i = 0; i < 2; ++i) {
      const int u = i * 128 + tid;           // 256 x 8-half units
      const int key = u >> 3, d8 = u & 7;
      v8h hv = *(const v8h*)(V + (size_t)(n * T + kb0 + key) * D + h * HD + d8 * 8);
#pragma unroll
      for (int jj = 0; jj < 8; ++jj) VsT[d8 * 8 + jj][key] = hv[jj];
    }
    wait_async0();
    __syncthreads();

    // scores: S(16x32) = Q(16x64) @ K^T(64x32), two 16-col WMMA tiles
    v8f s[2]; s[0] = {}; s[1] = {};
#pragma unroll
    for (int t = 0; t < 2; ++t) {
#pragma unroll
      for (int c = 0; c < 2; ++c) {
        v16h bk;
        const int col = t * 16 + rlo;  // key index within tile
#pragma unroll
        for (int i = 0; i < 8; ++i) {
          bk[i]     = Ks[col][c * 32 + kb + i];
          bk[8 + i] = Ks[col][c * 32 + kb + 16 + i];
        }
        s[t] = wmma16(qa[c], bk, s[t]);
      }
    }

    // online softmax (causal mask), row stats per accumulator register
#pragma unroll
    for (int r = 0; r < 8; ++r) {
      const int m    = r + 8 * (lane >> 4);
      const int grow = qbase + w * 16 + m;
      const int c0   = kb0 + rlo, c1 = c0 + 16;
      float x0 = (c0 <= grow) ? s[0][r] : -1e30f;
      float x1 = (c1 <= grow) ? s[1][r] : -1e30f;
      float tmax = fmaxf(x0, x1);
#pragma unroll
      for (int off = 1; off < 16; off <<= 1)
        tmax = fmaxf(tmax, __shfl_xor(tmax, off, 32));
      const float mnew = fmaxf(mrow[r], tmax);
      const float sc   = __expf(mrow[r] - mnew);
      const float p0   = __expf(x0 - mnew);
      const float p1   = __expf(x1 - mnew);
      float psum = p0 + p1;
#pragma unroll
      for (int off = 1; off < 16; off <<= 1)
        psum += __shfl_xor(psum, off, 32);
      lrow[r] = lrow[r] * sc + psum;
      mrow[r] = mnew;
#pragma unroll
      for (int t = 0; t < 4; ++t) accO[t][r] *= sc;
      Ps[w][m][rlo]      = (_Float16)p0;   // C-layout -> LDS
      Ps[w][m][rlo + 16] = (_Float16)p1;
    }

    // PV: accO(16x64) += P(16x32) @ V(32x64); per-wave LDS, DS is in-order
    v16h pa;
#pragma unroll
    for (int i = 0; i < 8; ++i) {
      pa[i]     = Ps[w][rlo][kb + i];
      pa[8 + i] = Ps[w][rlo][kb + 16 + i];
    }
#pragma unroll
    for (int t = 0; t < 4; ++t) {
      v16h bv;
      const int col = t * 16 + rlo;  // output dim
#pragma unroll
      for (int i = 0; i < 8; ++i) {
        bv[i]     = VsT[col][kb + i];
        bv[8 + i] = VsT[col][kb + 16 + i];
      }
      accO[t] = wmma16(pa, bv, accO[t]);
    }
    __syncthreads();
  }

  // normalize and store (f16)
#pragma unroll
  for (int t = 0; t < 4; ++t)
#pragma unroll
    for (int r = 0; r < 8; ++r) {
      const int m    = r + 8 * (lane >> 4);
      const int grow = qbase + w * 16 + m;
      O[(size_t)(n * T + grow) * D + h * HD + t * 16 + rlo] =
          (_Float16)(accO[t][r] / lrow[r]);
    }
}

// ---------------------------------------------------------------------------
// y = LayerNorm(x + res) * g + b.  D=1024, one block (256 thr) per row.
// resDiv!=0: res has one row per batch element (broadcast over sequence).
// ---------------------------------------------------------------------------
__global__ __launch_bounds__(256) void add_ln_kernel(
    const float* __restrict__ x, const float* __restrict__ res,
    const float* __restrict__ g, const float* __restrict__ b,
    float* __restrict__ y, int resDiv)
{
  const int D = 1024;
  __shared__ float red[256];
  const int row = blockIdx.x;
  const int tid = threadIdx.x;
  const float* xr = x + (size_t)row * D;
  const float* rr = res + (size_t)(resDiv ? (row >> 9) : row) * D;

  fx4 xl = *(const fx4*)(xr + tid * 4);
  fx4 rl = *(const fx4*)(rr + tid * 4);
  fx4 local = xl + rl;
  float s = local[0] + local[1] + local[2] + local[3];
  red[tid] = s; __syncthreads();
  for (int o = 128; o > 0; o >>= 1) { if (tid < o) red[tid] += red[tid + o]; __syncthreads(); }
  const float mean = red[0] * (1.0f / D);
  __syncthreads();
  fx4 dl = local - mean;
  float v = dl[0]*dl[0] + dl[1]*dl[1] + dl[2]*dl[2] + dl[3]*dl[3];
  red[tid] = v; __syncthreads();
  for (int o = 128; o > 0; o >>= 1) { if (tid < o) red[tid] += red[tid + o]; __syncthreads(); }
  const float rstd = rsqrtf(red[0] * (1.0f / D) + 1e-5f);
  fx4 gl = *(const fx4*)(g + tid * 4);
  fx4 bl = *(const fx4*)(b + tid * 4);
  fx4 out = gl * (dl * rstd) + bl;
  *(fx4*)(y + (size_t)row * D + tid * 4) = out;
}

// x[row] = tok_emb[captions[row]] + pos_emb[row % T]
__global__ __launch_bounds__(256) void embed_kernel(
    const int* __restrict__ cap, const float* __restrict__ tok,
    const float* __restrict__ pos, float* __restrict__ x)
{
  const int D = 1024, T = 512;
  const int row = blockIdx.x;
  const int t   = row & (T - 1);
  const int token = cap[row];
  const int tid = threadIdx.x;
  fx4 a = *(const fx4*)(tok + (size_t)token * D + tid * 4);
  fx4 p = *(const fx4*)(pos + (size_t)t * D + tid * 4);
  *(fx4*)(x + (size_t)row * D + tid * 4) = a + p;
}

// fp32 -> f16 weight pre-conversion; n % 2048 == 0, 8 elems/thread
__global__ __launch_bounds__(256) void cvt_f16_kernel(
    const float* __restrict__ s, _Float16* __restrict__ d)
{
  const size_t i = ((size_t)blockIdx.x * 256 + threadIdx.x) * 8;
  fx4 a = *(const fx4*)(s + i);
  fx4 b = *(const fx4*)(s + i + 4);
  *(v4h*)(d + i)     = __builtin_convertvector(a, v4h);
  *(v4h*)(d + i + 4) = __builtin_convertvector(b, v4h);
}

extern "C" void kernel_launch(void* const* d_in, const int* in_sizes, int n_in,
                              void* d_out, int out_size, void* d_ws, size_t ws_size,
                              hipStream_t stream) {
  (void)in_sizes; (void)n_in; (void)out_size; (void)ws_size;
  const int NB = 8, T = 512, Dm = 1024, FFN = 2048, VOC = 32000, DIN = 2048;
  const int NT = NB * T, L = 4;

  const float* features = (const float*)d_in[0];
  const int*   captions = (const int*)  d_in[1];
  const float* tok_emb  = (const float*)d_in[2];
  const float* pos_emb  = (const float*)d_in[3];
  const float* feat_w   = (const float*)d_in[4];
  const float* feat_b   = (const float*)d_in[5];
  const float* out_w    = (const float*)d_in[6];
  const float* out_b    = (const float*)d_in[7];
  // per-layer dict order: sa qw,kw,vw,ow,qb,kb,vb,ob | ca qw..ob | ffw1,ffb1,ffw2,ffb2 | n1w..n3b
  auto lp = [&](int layer, int idx) -> const float* {
    return (const float*)d_in[8 + layer * 26 + idx];
  };

  // ---- workspace layout (byte cursor, 256B aligned chunks) ----
  char* base = (char*)d_ws;
  size_t off = 0;
  auto allocF = [&](size_t n) { float* p = (float*)(base + off);
                                off = (off + n * 4 + 255) & ~(size_t)255; return p; };
  auto allocH = [&](size_t n) { _Float16* p = (_Float16*)(base + off);
                                off = (off + n * 2 + 255) & ~(size_t)255; return p; };

  const size_t SZ = (size_t)NT * Dm;       // 4 Mi elems
  float*    xA   = allocF(SZ);
  float*    xB   = allocF(SZ);
  float*    t1   = allocF(SZ);
  _Float16* q16  = allocH(SZ);
  _Float16* k16  = allocH(SZ);
  _Float16* v16  = allocH(SZ);
  _Float16* att16= allocH(SZ);
  _Float16* hff16= q16;                     // NTxFFN f16 reuses q16+k16 region
  float*    cond = allocF((size_t)NB * Dm);
  float*    cv   = allocF((size_t)NB * Dm);
  float*    ca   = allocF((size_t)NB * Dm);

  // f16 weight copies
  _Float16* w_feat = allocH((size_t)DIN * Dm);
  _Float16* w_out  = allocH((size_t)Dm * VOC);
  _Float16 *w_sa[4][4], *w_cav[4], *w_cao[4], *w_ff1[4], *w_ff2[4];
  for (int l = 0; l < L; ++l) {
    for (int i = 0; i < 4; ++i) w_sa[l][i] = allocH((size_t)Dm * Dm);
    w_cav[l] = allocH((size_t)Dm * Dm);
    w_cao[l] = allocH((size_t)Dm * Dm);
    w_ff1[l] = allocH((size_t)Dm * FFN);
    w_ff2[l] = allocH((size_t)FFN * Dm);
  }

  auto cvt = [&](const float* s, _Float16* d, size_t n) {
    cvt_f16_kernel<<<(uint32_t)(n / 2048), 256, 0, stream>>>(s, d);
  };
  cvt(feat_w, w_feat, (size_t)DIN * Dm);
  cvt(out_w,  w_out,  (size_t)Dm * VOC);
  for (int l = 0; l < L; ++l) {
    for (int i = 0; i < 4; ++i) cvt(lp(l, i), w_sa[l][i], (size_t)Dm * Dm);
    cvt(lp(l, 10), w_cav[l], (size_t)Dm * Dm);
    cvt(lp(l, 11), w_cao[l], (size_t)Dm * Dm);
    cvt(lp(l, 16), w_ff1[l], (size_t)Dm * FFN);
    cvt(lp(l, 18), w_ff2[l], (size_t)FFN * Dm);
  }

  // GEMM launchers: A-f32/out-f32, A-f32/out-f16, A-f16/out-f32
  auto gemm_ff = [&](const float* A, const _Float16* B, const float* bias, float* C,
                     int M, int N, int K, int relu, float sc) {
    hipLaunchKernelGGL((gemm_wmma_kernel<false, false>),
                       dim3(N / 128, (M + 127) / 128), dim3(256), 0, stream,
                       (const void*)A, B, bias, (void*)C, M, N, K, relu, sc);
  };
  auto gemm_fh = [&](const float* A, const _Float16* B, const float* bias, _Float16* C,
                     int M, int N, int K, int relu, float sc) {
    hipLaunchKernelGGL((gemm_wmma_kernel<false, true>),
                       dim3(N / 128, (M + 127) / 128), dim3(256), 0, stream,
                       (const void*)A, B, bias, (void*)C, M, N, K, relu, sc);
  };
  auto gemm_hf = [&](const _Float16* A, const _Float16* B, const float* bias, float* C,
                     int M, int N, int K, int relu, float sc) {
    hipLaunchKernelGGL((gemm_wmma_kernel<true, false>),
                       dim3(N / 128, (M + 127) / 128), dim3(256), 0, stream,
                       (const void*)A, B, bias, (void*)C, M, N, K, relu, sc);
  };

  embed_kernel<<<NT, 256, 0, stream>>>(captions, tok_emb, pos_emb, xA);
  gemm_ff(features, w_feat, feat_b, cond, NB, Dm, DIN, 0, 1.0f);

  float* x = xA; float* xo = xB;
  for (int l = 0; l < L; ++l) {
    // --- self attention (Q scaled by 1/sqrt(head_dim) in its epilogue) ---
    gemm_fh(x, w_sa[l][0], lp(l, 4), q16, NT, Dm, Dm, 0, 0.125f);
    gemm_fh(x, w_sa[l][1], lp(l, 5), k16, NT, Dm, Dm, 0, 1.0f);
    gemm_fh(x, w_sa[l][2], lp(l, 6), v16, NT, Dm, Dm, 0, 1.0f);
    attn_kernel<<<dim3(T / 64, 16, NB), 128, 0, stream>>>(q16, k16, v16, att16);
    gemm_hf(att16, w_sa[l][3], lp(l, 7), t1, NT, Dm, Dm, 0, 1.0f);
    add_ln_kernel<<<NT, 256, 0, stream>>>(x, t1, lp(l, 20), lp(l, 21), xo, 0);
    { float* tmp = x; x = xo; xo = tmp; }

    // --- cross attention: single KV token => softmax==1 => V/O proj broadcast ---
    gemm_ff(cond, w_cav[l], lp(l, 14), cv, NB, Dm, Dm, 0, 1.0f);
    gemm_ff(cv,   w_cao[l], lp(l, 15), ca, NB, Dm, Dm, 0, 1.0f);
    add_ln_kernel<<<NT, 256, 0, stream>>>(x, ca, lp(l, 22), lp(l, 23), xo, 1);
    { float* tmp = x; x = xo; xo = tmp; }

    // --- FFN ---
    gemm_fh(x,     w_ff1[l], lp(l, 17), hff16, NT, FFN, Dm, 1, 1.0f);  // ReLU fused
    gemm_hf(hff16, w_ff2[l], lp(l, 19), t1,    NT, Dm, FFN, 0, 1.0f);
    add_ln_kernel<<<NT, 256, 0, stream>>>(x, t1, lp(l, 24), lp(l, 25), xo, 0);
    { float* tmp = x; x = xo; xo = tmp; }
  }

  // logits: dominant GEMM, 4096x32000x1024, f16 weights in L2
  gemm_ff(x, w_out, out_b, (float*)d_out, NT, VOC, Dm, 0, 1.0f);
}
